// SpikingLayer_6871947674303
// MI455X (gfx1250) — matrix operand order repaired and checked by reading, not compiled, not run on hardware
//
#include <hip/hip_runtime.h>
#include <hip/hip_bf16.h>
#include <cstddef>
#include <cstdint>

// ---------------------------------------------------------------------------
// Types for CDNA5 WMMA (wave32): bf16 A/B fragments (16 x bf16), f32 C/D (8 x f32)
// ---------------------------------------------------------------------------
typedef __bf16 bf16_t;
typedef bf16_t v16bf __attribute__((ext_vector_type(16)));
typedef float  v8f   __attribute__((ext_vector_type(8)));
typedef int    v4i   __attribute__((vector_size(16)));   // matches async-LDS builtin param
typedef __attribute__((address_space(1))) v4i* g_v4i_p;  // global (AS1) int4*
typedef __attribute__((address_space(3))) v4i* s_v4i_p;  // LDS (AS3) int4*

#define B_    32
#define T_    1024
#define DIN_  512
#define H_    1024
#define G4_   4096
#define KTOT_ 1536              // DIN + H (fused [x_t ; h_t] K dimension)
#define NGRP_ 2                 // batch groups of 16 rows (M=16 per WMMA)
#define WGPG_ 8                 // cooperating workgroups per group (128 h-cols each)
#define NW_   8                 // waves per workgroup (256 threads)
#define KT_W  (KTOT_/32)        // 48 K-tiles in packed recurrent weights
#define NT_W  (G4_/16)          // 256 N-tiles (= 64 h-tiles x 4 gates)
#define KT_F  (H_/32)           // 32 K-tiles in packed fc weights
#define NT_F  (512/16)          // 32 N-tiles
#define ASTRIDE_ 1544           // LDS A-tile row stride (1536 + 8 pad vs bank conflicts)

// CDNA5 async global->LDS path (ASYNCcnt); fall back to plain LDS copies.
#if __has_builtin(__builtin_amdgcn_global_load_async_to_lds_b128) && \
    __has_builtin(__builtin_amdgcn_s_wait_asynccnt)
#define HAVE_ASYNC_LDS 1
#else
#define HAVE_ASYNC_LDS 0
#endif

// ---------------------------------------------------------------------------
// Workspace layout (bytes, all 256B aligned)
// ---------------------------------------------------------------------------
static constexpr size_t SZ_XB  = (size_t)B_*T_*DIN_*2;        // x in bf16        33.5 MB
static constexpr size_t SZ_WP  = (size_t)NT_W*KT_W*512*2;     // packed [w_ih;w_hh] 12.6 MB
static constexpr size_t SZ_FP  = (size_t)NT_F*KT_F*512*2;     // packed fc_w        1 MB
static constexpr size_t SZ_BS  = (size_t)G4_*4;               // b_ih+b_hh f32
static constexpr size_t SZ_SPK = (size_t)B_*T_*H_*2;          // spikes bf16       67 MB
static constexpr size_t SZ_HS  = (size_t)2*NGRP_*16*H_*2;     // double-buffered h
static constexpr size_t OFF_XB  = 0;
static constexpr size_t OFF_WP  = OFF_XB  + SZ_XB;
static constexpr size_t OFF_FP  = OFF_WP  + SZ_WP;
static constexpr size_t OFF_BS  = OFF_FP  + SZ_FP;
static constexpr size_t OFF_SPK = OFF_BS  + SZ_BS;
static constexpr size_t OFF_HS  = OFF_SPK + SZ_SPK;
static constexpr size_t OFF_BAR = OFF_HS  + SZ_HS;            // 64 ints of barrier state

// ---------------------------------------------------------------------------
// Fragment loaders (per-lane contiguous 16B loads -> global/ds _load_b128)
// ---------------------------------------------------------------------------
__device__ __forceinline__ v16bf ld16x2(const bf16_t* p0, const bf16_t* p1) {
  union { uint4 v; bf16_t h[8]; } u0, u1;
  u0.v = *reinterpret_cast<const uint4*>(p0);
  u1.v = *reinterpret_cast<const uint4*>(p1);
  v16bf r;
#pragma unroll
  for (int i = 0; i < 8; ++i) { r[i] = u0.h[i]; r[8 + i] = u1.h[i]; }
  return r;
}
// A 16x32 bf16: lanes 0-15 row m=lane, K {0..15}; lanes 16-31 K {16..31}
__device__ __forceinline__ v16bf ld_a(const bf16_t* row_k0, int hi) {
  return ld16x2(row_k0 + hi * 16, row_k0 + hi * 16 + 8);
}
// B 32x16 bf16: pre-packed so each lane reads 32 contiguous bytes
__device__ __forceinline__ v16bf ld_b(const bf16_t* p) { return ld16x2(p, p + 8); }

// Hardware transcendental tanh (V_TANH_F32 on gfx1250) when available.
#if __has_builtin(__builtin_amdgcn_tanhf)
__device__ __forceinline__ float fast_tanh(float x) { return __builtin_amdgcn_tanhf(x); }
#else
__device__ __forceinline__ float fast_tanh(float x) { return tanhf(x); }
#endif
// sigmoid(x) = 0.5*(1 + tanh(x/2))  (exact identity -> one v_tanh each)
__device__ __forceinline__ float sigf(float x) { return 0.5f * fast_tanh(0.5f * x) + 0.5f; }

// ---------------------------------------------------------------------------
// One-shot prep kernels
// ---------------------------------------------------------------------------
__global__ void init_bar_kernel(int* bar) { if (threadIdx.x < 64) bar[threadIdx.x] = 0; }

__global__ void convert_x_kernel(const float* __restrict__ x, bf16_t* __restrict__ xb, size_t n) {
  size_t i = (size_t)blockIdx.x * blockDim.x + threadIdx.x;
  if (i < n) xb[i] = (bf16_t)x[i];
}

__global__ void bias_sum_kernel(const float* __restrict__ b_ih, const float* __restrict__ b_hh,
                                float* __restrict__ bs) {
  int i = blockIdx.x * blockDim.x + threadIdx.x;
  if (i < G4_) bs[i] = b_ih[i] + b_hh[i];
}

// Pack combined [w_ih ; w_hh] (K=1536, N=4096) into per-lane B-fragment order,
// gate-interleaved: blob = ((hTile*KT_W + kT)*4 + gate) so the 4 gate fragments
// of one (hTile,kT) are 4 contiguous 1KB blobs (single base + imm offsets).
__global__ void pack_w_kernel(const float* __restrict__ w_ih, const float* __restrict__ w_hh,
                              bf16_t* __restrict__ wp) {
  size_t idx = (size_t)blockIdx.x * blockDim.x + threadIdx.x;
  if (idx >= (size_t)NT_W * KT_W * 512) return;
  int i    = (int)(idx & 15);
  int lane = (int)((idx >> 4) & 31);
  size_t blob = idx >> 9;
  int gate  = (int)(blob & 3);
  size_t kh = blob >> 2;
  int kT    = (int)(kh % KT_W);
  int hTile = (int)(kh / KT_W);
  int n = gate * H_ + hTile * 16 + (lane & 15);
  int k = kT * 32 + (lane >> 4) * 16 + i;
  float v = (k < DIN_) ? w_ih[(size_t)n * DIN_ + k] : w_hh[(size_t)n * H_ + (k - DIN_)];
  wp[idx] = (bf16_t)v;
}

__global__ void pack_fcw_kernel(const float* __restrict__ fc_w, bf16_t* __restrict__ fp) {
  size_t idx = (size_t)blockIdx.x * blockDim.x + threadIdx.x;
  if (idx >= (size_t)NT_F * KT_F * 512) return;
  int i    = (int)(idx & 15);
  int lane = (int)((idx >> 4) & 31);
  size_t blob = idx >> 9;
  int kT = (int)(blob % KT_F);
  int nT = (int)(blob / KT_F);
  int n = nT * 16 + (lane & 15);
  int k = kT * 32 + (lane >> 4) * 16 + i;
  fp[idx] = (bf16_t)fc_w[(size_t)n * H_ + k];
}

// ---------------------------------------------------------------------------
// Device-scope per-group barrier (arrive/release with monotonically rising phase)
// ---------------------------------------------------------------------------
__device__ __forceinline__ void group_barrier(int* bar, int g, int expect) {
  __syncthreads();
  if (threadIdx.x == 0) {
    __threadfence();
    int* arr = bar + g * 32;
    int* rel = bar + g * 32 + 1;
    int prev = __hip_atomic_fetch_add(arr, 1, __ATOMIC_ACQ_REL, __HIP_MEMORY_SCOPE_AGENT);
    if (prev == WGPG_ - 1) {
      __hip_atomic_store(arr, 0, __ATOMIC_RELAXED, __HIP_MEMORY_SCOPE_AGENT);
      __hip_atomic_fetch_add(rel, 1, __ATOMIC_RELEASE, __HIP_MEMORY_SCOPE_AGENT);
    }
    while (__hip_atomic_load(rel, __ATOMIC_ACQUIRE, __HIP_MEMORY_SCOPE_AGENT) < expect)
      __builtin_amdgcn_s_sleep(1);
  }
  __syncthreads();
}

// ---------------------------------------------------------------------------
// Phase 2: fused LSTM recurrence.
//   2 groups x 8 WGs x 8 waves; per step the WG stages A=[x_t;h_t] (16x1536 bf16)
//   into LDS via async-to-LDS, then each wave runs the K-loop of
//   v_wmma_f32_16x16x32_bf16 for its 4 gate tiles (8 B-loads in flight, one base
//   pointer + imm offsets), f32 LSTM pointwise via v_tanh_f32, h double-buffered
//   in L2, device barrier per step.
// ---------------------------------------------------------------------------
__global__ void lstm_recurrence_kernel(const bf16_t* __restrict__ xb,
                                       const bf16_t* __restrict__ wp,
                                       const float*  __restrict__ bsum,
                                       const float*  __restrict__ syn0,
                                       const float*  __restrict__ mem0,
                                       const float*  __restrict__ thr_p,
                                       bf16_t* __restrict__ spks,
                                       bf16_t* hsh,
                                       int* bar,
                                       float* __restrict__ syn_out,
                                       float* __restrict__ mem_out) {
  __shared__ __attribute__((aligned(16))) bf16_t a_lds[16 * ASTRIDE_];

  const int tid    = threadIdx.x;
  const int lane   = tid & 31;
  const int wave   = tid >> 5;
  const int laneLo = lane & 15;
  const int hi     = lane >> 4;
  const int g      = blockIdx.x / WGPG_;
  const int wgi    = blockIdx.x % WGPG_;
  const int c0     = wgi * 128 + wave * 16;     // owned h-column base
  const int hTile  = c0 >> 4;
  const float thr  = *thr_p;

  // Persistent f32 cell/hidden fragments (rows m=e+8*hi, col c0+laneLo)
  v8f c_frag, h_frag;
  bf16_t* hbuf0 = hsh + (size_t)g * 16 * H_;    // parity-0 buffer for this group
#pragma unroll
  for (int e = 0; e < 8; ++e) {
    int row = e + 8 * hi;
    int bb  = g * 16 + row;
    int col = c0 + laneLo;
    c_frag[e] = syn0[(size_t)bb * H_ + col];
    h_frag[e] = mem0[(size_t)bb * H_ + col];
    hbuf0[(size_t)row * H_ + col] = (bf16_t)h_frag[e];
  }

  int phase = 0;
  group_barrier(bar, g, ++phase);               // initial h visible group-wide

  // Per-wave packed-weight base: 4 gate blobs contiguous per (hTile,kT)
  const bf16_t* wbase = wp + ((size_t)hTile * KT_W * 4) * 512 + lane * 16;

  for (int t = 0; t < T_; ++t) {
    const bf16_t* hbuf_r = hsh + ((size_t)((t    ) & 1) * NGRP_ + g) * 16 * H_;
    bf16_t*       hbuf_w = hsh + ((size_t)((t + 1) & 1) * NGRP_ + g) * 16 * H_;

    // --- Stage A = [x_t ; h_t] (16 rows x 1536) into LDS, once per WG ---
    // 3072 chunks of 16B; 12 per thread; consecutive tids -> consecutive 16B.
#pragma unroll
    for (int c = 0; c < 12; ++c) {
      int chunk = tid + c * 256;
      int r   = chunk / 192;
      int col = (chunk % 192) * 8;
      const bf16_t* src = (col < DIN_)
          ? (xb + ((size_t)(g * 16 + r) * T_ + t) * DIN_ + col)
          : (hbuf_r + (size_t)r * H_ + (col - DIN_));
      bf16_t* dst = &a_lds[r * ASTRIDE_ + col];
#if HAVE_ASYNC_LDS
      __builtin_amdgcn_global_load_async_to_lds_b128(
          (g_v4i_p)src, (s_v4i_p)dst, 0, 0);
#else
      *(uint4*)dst = *(const uint4*)src;
#endif
    }
#if HAVE_ASYNC_LDS
    __builtin_amdgcn_s_wait_asynccnt(0);
#endif
    __syncthreads();

    // --- Gate accumulators, bias-initialized (broadcast across rows) ---
    v8f acc[4];
#pragma unroll
    for (int gate = 0; gate < 4; ++gate) {
      float bv = bsum[gate * H_ + c0 + laneLo];
#pragma unroll
      for (int e = 0; e < 8; ++e) acc[gate][e] = bv;
    }

    // --- K-loop: A from LDS (ds_load_b128), B from L2; 8 B-loads in flight ---
    const bf16_t* arow = &a_lds[(size_t)laneLo * ASTRIDE_];
#pragma unroll 2
    for (int k0 = 0; k0 < KTOT_; k0 += 32) {
      const bf16_t* bp = wbase + (size_t)(k0 >> 5) * 4 * 512;
      __builtin_prefetch(bp + 4 * 512, 0, 1);   // next k-tile (4KB ahead)
      v16bf a  = ld_a(arow + k0, hi);
      v16bf b0 = ld_b(bp);
      v16bf b1 = ld_b(bp + 512);
      v16bf b2 = ld_b(bp + 1024);
      v16bf b3 = ld_b(bp + 1536);
      acc[0] = __builtin_amdgcn_wmma_f32_16x16x32_bf16(false, a, false, b0, (short)0, acc[0], false, false);
      acc[1] = __builtin_amdgcn_wmma_f32_16x16x32_bf16(false, a, false, b1, (short)0, acc[1], false, false);
      acc[2] = __builtin_amdgcn_wmma_f32_16x16x32_bf16(false, a, false, b2, (short)0, acc[2], false, false);
      acc[3] = __builtin_amdgcn_wmma_f32_16x16x32_bf16(false, a, false, b3, (short)0, acc[3], false, false);
    }

    // --- LSTM pointwise (f32, v_tanh_f32) + Heaviside spike ---
#pragma unroll
    for (int e = 0; e < 8; ++e) {
      float iv = sigf(acc[0][e]);
      float fv = sigf(acc[1][e]);
      float gv = fast_tanh(acc[2][e]);
      float ov = sigf(acc[3][e]);
      float cn = fv * c_frag[e] + iv * gv;
      float hn = ov * fast_tanh(cn);
      c_frag[e] = cn;
      h_frag[e] = hn;
      int row = e + 8 * hi;
      int col = c0 + laneLo;
      hbuf_w[(size_t)row * H_ + col] = (bf16_t)hn;
      float sp = (hn - thr) > 0.0f ? 1.0f : 0.0f;
      spks[((size_t)(g * 16 + row) * T_ + t) * H_ + col] = (bf16_t)sp;
    }

    group_barrier(bar, g, ++phase);             // h_{t+1} visible; buffers swap
  }

  // Final cell / hidden state -> tuple outputs (syn, mem)
#pragma unroll
  for (int e = 0; e < 8; ++e) {
    int row = e + 8 * hi;
    int bb  = g * 16 + row;
    int col = c0 + laneLo;
    syn_out[(size_t)bb * H_ + col] = c_frag[e];
    mem_out[(size_t)bb * H_ + col] = h_frag[e];
  }
}

// ---------------------------------------------------------------------------
// Phase 3: y = spks @ fc_w^T + fc_b ; out = gamma*tanh(alpha*y)+beta
// One 16x16 output tile per wave; K=1024 loop of bf16 WMMA.
// ---------------------------------------------------------------------------
__global__ void fc_dyt_kernel(const bf16_t* __restrict__ spks,
                              const bf16_t* __restrict__ fp,
                              const float* __restrict__ fcb,
                              const float* __restrict__ alpha_p,
                              const float* __restrict__ gamma,
                              const float* __restrict__ beta,
                              float* __restrict__ out) {
  const int lane   = threadIdx.x & 31;
  const int wave   = threadIdx.x >> 5;
  const int laneLo = lane & 15;
  const int hi     = lane >> 4;
  const int tile   = blockIdx.x * NW_ + wave;
  const int mT     = tile / NT_F;
  const int nT     = tile % NT_F;
  const float alpha = *alpha_p;

  const bf16_t* arow  = spks + (size_t)(mT * 16 + laneLo) * H_;
  const bf16_t* bbase = fp + (size_t)nT * KT_F * 512 + lane * 16;
  v8f acc = {};
#pragma unroll 2
  for (int k0 = 0; k0 < H_; k0 += 32) {
    const bf16_t* bp = bbase + (size_t)(k0 >> 5) * 512;
    __builtin_prefetch(bp + 512, 0, 1);
    v16bf a = ld_a(arow + k0, hi);
    v16bf b = ld_b(bp);
    acc = __builtin_amdgcn_wmma_f32_16x16x32_bf16(
        false, a, false, b, (short)0, acc, false, false);
  }
  int n = nT * 16 + laneLo;
  float bb = fcb[n], ga = gamma[n], be = beta[n];
#pragma unroll
  for (int e = 0; e < 8; ++e) {
    int mr = mT * 16 + e + 8 * hi;
    float y = acc[e] + bb;
    out[(size_t)mr * 512 + n] = ga * fast_tanh(alpha * y) + be;
  }
}

// ---------------------------------------------------------------------------
// Host launch
// ---------------------------------------------------------------------------
extern "C" void kernel_launch(void* const* d_in, const int* in_sizes, int n_in,
                              void* d_out, int out_size, void* d_ws, size_t ws_size,
                              hipStream_t stream) {
  (void)in_sizes; (void)n_in; (void)out_size; (void)ws_size;
  const float* x      = (const float*)d_in[0];
  const float* syn0   = (const float*)d_in[1];
  const float* mem0   = (const float*)d_in[2];
  const float* w_ih   = (const float*)d_in[3];
  const float* w_hh   = (const float*)d_in[4];
  const float* b_ih   = (const float*)d_in[5];
  const float* b_hh   = (const float*)d_in[6];
  const float* thr    = (const float*)d_in[7];
  const float* fc_w   = (const float*)d_in[8];
  const float* fc_b   = (const float*)d_in[9];
  const float* alpha  = (const float*)d_in[10];
  const float* gamma  = (const float*)d_in[11];
  const float* beta   = (const float*)d_in[12];

  char* ws = (char*)d_ws;
  bf16_t* xb   = (bf16_t*)(ws + OFF_XB);
  bf16_t* wp   = (bf16_t*)(ws + OFF_WP);
  bf16_t* fp   = (bf16_t*)(ws + OFF_FP);
  float*  bs   = (float*) (ws + OFF_BS);
  bf16_t* spks = (bf16_t*)(ws + OFF_SPK);
  bf16_t* hsh  = (bf16_t*)(ws + OFF_HS);
  int*    bar  = (int*)   (ws + OFF_BAR);

  float* out_y   = (float*)d_out;                         // [B,T,512]
  float* syn_out = out_y + (size_t)B_ * T_ * 512;         // [B,H]
  float* mem_out = syn_out + (size_t)B_ * H_;             // [B,H]

  // Phase 0: prep
  init_bar_kernel<<<1, 64, 0, stream>>>(bar);
  {
    size_t n = (size_t)B_ * T_ * DIN_;
    convert_x_kernel<<<(unsigned)((n + 255) / 256), 256, 0, stream>>>(x, xb, n);
  }
  pack_w_kernel<<<(NT_W * KT_W * 512) / 256, 256, 0, stream>>>(w_ih, w_hh, wp);
  pack_fcw_kernel<<<(NT_F * KT_F * 512) / 256, 256, 0, stream>>>(fc_w, fp);
  bias_sum_kernel<<<G4_ / 256, 256, 0, stream>>>(b_ih, b_hh, bs);

  // Phase 2: sequential LSTM recurrence (16 cooperating workgroups)
  lstm_recurrence_kernel<<<NGRP_ * WGPG_, NW_ * 32, 0, stream>>>(
      xb, wp, bs, syn0, mem0, thr, spks, hsh, bar, syn_out, mem_out);

  // Phase 3: spike projection + DyT
  fc_dyt_kernel<<<(2048 * NT_F) / NW_, NW_ * 32, 0, stream>>>(
      spks, fp, fc_b, alpha, gamma, beta, out_y);
}
